// TransformerLayer_28656021799491
// MI455X (gfx1250) — matrix operand order repaired and checked by reading, not compiled, not run on hardware
//
#include <hip/hip_runtime.h>
#include <hip/hip_bf16.h>
#include <math.h>

typedef _Float16 h16;
typedef __attribute__((ext_vector_type(16))) _Float16 v16h;
typedef __attribute__((ext_vector_type(8)))  float    v8f;

static constexpr int BB = 512;
static constexpr int S_ = 127;
static constexpr int L  = 128;
static constexpr int D  = 512;
static constexpr int H_ = 8;
static constexpr int DH = 64;
static constexpr int FF = 1024;
static constexpr int NT = BB * L;   // 65536 token rows

// ---------------------------------------------------------------- WMMA helpers
__device__ __forceinline__ v8f wmma_f16(v16h a, v16h b, v8f c) {
  return __builtin_amdgcn_wmma_f32_16x16x32_f16(false, a, false, b, (short)0, c,
                                                false, false);
}

// A-matrix fragment (16x32, M x K), row-major source with leading dim lda.
// Lane l: row = m0 + (l&15); element j = K kb + 16*(l>>4) + j (contiguous).
__device__ __forceinline__ v16h load_a_frag(const h16* A, int lda, int m0,
                                            int kb, int lane) {
  const int row   = m0 + (lane & 15);
  const int khalf = lane >> 4;
  const h16* p = A + (size_t)row * lda + kb + khalf * 16;
  union { v16h v; uint4 u[2]; } t;
  t.u[0] = *(const uint4*)(p);
  t.u[1] = *(const uint4*)(p + 8);
  return t.v;
}

// B-matrix fragment (32x16, K x N) where B[k][n] = W[n][k], W row-major (N,K).
// Lane l: col n = n0 + (l&15); element j = K kb + 16*(l>>4) + j (contiguous).
__device__ __forceinline__ v16h load_b_frag(const h16* W, int ldw, int n0,
                                            int kb, int lane) {
  const int n     = n0 + (lane & 15);
  const int khalf = lane >> 4;
  return *(const v16h*)(W + (size_t)n * ldw + kb + khalf * 16);
}

// ---------------------------------------------------------------- prep kernels
__global__ void __launch_bounds__(256)
pack_tokens_kernel(const float* __restrict__ node, const float* __restrict__ edge,
                   float* __restrict__ tok32, h16* __restrict__ tok16) {
  const size_t total = (size_t)NT * D;
  for (size_t i = blockIdx.x * 256ull + threadIdx.x; i < total;
       i += (size_t)gridDim.x * 256ull) {
    const int d = (int)(i & (D - 1));
    const int m = (int)(i >> 9);
    const int b = m >> 7;
    const int l = m & (L - 1);
    float v = (l == 0) ? node[(size_t)b * D + d]
                       : edge[((size_t)b * S_ + (l - 1)) * D + d];
    tok32[i] = v;
    tok16[i] = (h16)v;
  }
}

__global__ void __launch_bounds__(256)
cvt_f32_to_f16_kernel(const float* __restrict__ src, h16* __restrict__ dst, int n) {
  for (int i = blockIdx.x * 256 + threadIdx.x; i < n; i += gridDim.x * 256)
    dst[i] = (h16)src[i];
}

// ---------------------------------------------------------------- GEMM 1: QKV
// C = tok16(NT x 512) @ w_qkv^T(512 x 1536) + b ; scatter f16 Q/K/V (B,H,L,DH)
// wave tile: 32M x 64N; block: 8 waves = 128M x 128N
__global__ void __launch_bounds__(256)
gemm_qkv_kernel(const h16* __restrict__ A, const h16* __restrict__ W,
                const float* __restrict__ bias, h16* __restrict__ Q,
                h16* __restrict__ K_, h16* __restrict__ V) {
  const int lane = threadIdx.x & 31;
  const int w    = threadIdx.x >> 5;
  const int m0 = blockIdx.x * 128 + (w & 3) * 32;
  const int n0 = blockIdx.y * 128 + (w >> 2) * 64;
  v8f acc[2][4] = {};
  for (int kb = 0; kb < D; kb += 32) {
    v16h a0 = load_a_frag(A, D, m0, kb, lane);
    v16h a1 = load_a_frag(A, D, m0 + 16, kb, lane);
    if (kb + 64 < D) {
      __builtin_prefetch(A + (size_t)(m0 + (lane & 15)) * D + kb + 64, 0, 3);
      __builtin_prefetch(A + (size_t)(m0 + 16 + (lane & 15)) * D + kb + 64, 0, 3);
    }
#pragma unroll
    for (int t = 0; t < 4; ++t) {
      v16h b = load_b_frag(W, D, n0 + 16 * t, kb, lane);
      acc[0][t] = wmma_f16(a0, b, acc[0][t]);
      acc[1][t] = wmma_f16(a1, b, acc[1][t]);
    }
  }
  const int mhalf = lane >> 4, nloc = lane & 15;
#pragma unroll
  for (int t = 0; t < 4; ++t) {
    const int n    = n0 + 16 * t + nloc;
    const int sect = n >> 9;                 // 0=q 1=k 2=v
    const int h    = (n >> 6) & 7;
    const int dh   = n & 63;
    const float bs = bias[n];
    const float sc = (sect == 0) ? 0.125f : 1.0f;   // fold 1/sqrt(DH) into Q
    h16* dst = (sect == 0) ? Q : ((sect == 1) ? K_ : V);
#pragma unroll
    for (int mt = 0; mt < 2; ++mt)
#pragma unroll
      for (int r = 0; r < 8; ++r) {
        const int m = m0 + mt * 16 + r + 8 * mhalf;
        const int b = m >> 7, l = m & (L - 1);
        dst[(((size_t)(b * H_ + h)) * L + l) * DH + dh] =
            (h16)((acc[mt][t][r] + bs) * sc);
      }
  }
}

// ---------------------------------------------------------------- attention
// One block per (b,h): S = Q K^T + log(clip(cutoff)); softmax; ctx = P V.
__global__ void __launch_bounds__(256)
attn_kernel(const h16* __restrict__ Q, const h16* __restrict__ K_,
            const h16* __restrict__ V, const float* __restrict__ cutoff,
            h16* __restrict__ ctx) {
  __shared__ h16 Vt[DH][L];        // V transposed: [dh][key], 16 KB
  __shared__ h16 Pb[8][16][L];     // per-wave probabilities, 32 KB
  const int bh = blockIdx.x;
  const int b  = bh >> 3;
  const int h  = bh & 7;
  const h16* Qb = Q + (size_t)bh * L * DH;
  const h16* Kb = K_ + (size_t)bh * L * DH;
  const h16* Vb = V + (size_t)bh * L * DH;

  for (int i = threadIdx.x; i < L * DH; i += 256) {
    const int key = i >> 6, dh = i & 63;
    Vt[dh][key] = Vb[i];
  }
  __syncthreads();

  const int lane = threadIdx.x & 31;
  const int w    = threadIdx.x >> 5;
  const int q0   = w * 16;
  const int mhalf = lane >> 4, nloc = lane & 15;

  // ---- scores: 16 query rows x 128 keys, kept in registers
  v16h qa0 = load_a_frag(Qb, DH, q0, 0, lane);
  v16h qa1 = load_a_frag(Qb, DH, q0, 32, lane);
  v8f st[8];
#pragma unroll
  for (int t = 0; t < 8; ++t) {
    v16h b0 = load_b_frag(Kb, DH, 16 * t, 0, lane);
    v16h b1 = load_b_frag(Kb, DH, 16 * t, 32, lane);
    v8f a = {};
    a = wmma_f16(qa0, b0, a);
    a = wmma_f16(qa1, b1, a);
    st[t] = a;
  }
  // ---- + log-mask
  const float* cf = cutoff + (size_t)b * L * L;
#pragma unroll
  for (int t = 0; t < 8; ++t)
#pragma unroll
    for (int r = 0; r < 8; ++r) {
      const int qi = q0 + r + 8 * mhalf;
      const int ki = 16 * t + nloc;
      st[t][r] += __logf(fmaxf(cf[qi * L + ki], 1e-15f));
    }
  // ---- row softmax (rows live in one 16-lane half; reduce with xor 1..8)
#pragma unroll
  for (int r = 0; r < 8; ++r) {
    float mx = -3.4e38f;
#pragma unroll
    for (int t = 0; t < 8; ++t) mx = fmaxf(mx, st[t][r]);
#pragma unroll
    for (int dd = 8; dd >= 1; dd >>= 1) mx = fmaxf(mx, __shfl_xor(mx, dd, 32));
    float sum = 0.f;
#pragma unroll
    for (int t = 0; t < 8; ++t) {
      float e = __expf(st[t][r] - mx);
      st[t][r] = e;
      sum += e;
    }
#pragma unroll
    for (int dd = 8; dd >= 1; dd >>= 1) sum += __shfl_xor(sum, dd, 32);
    const float inv = 1.0f / sum;
#pragma unroll
    for (int t = 0; t < 8; ++t) st[t][r] *= inv;
  }
  // ---- P -> LDS (reshape to A-fragment order)
#pragma unroll
  for (int t = 0; t < 8; ++t)
#pragma unroll
    for (int r = 0; r < 8; ++r)
      Pb[w][r + 8 * mhalf][16 * t + nloc] = (h16)st[t][r];
  __syncthreads();

  // ---- ctx = P (16x128) @ V (128x64)
  const h16* Pw = &Pb[w][0][0];
#pragma unroll
  for (int d = 0; d < 4; ++d) {
    v8f a = {};
#pragma unroll
    for (int kk = 0; kk < L; kk += 32) {
      v16h pa = load_a_frag(Pw, L, 0, kk, lane);
      v16h vb = *(const v16h*)(&Vt[16 * d + nloc][kk + mhalf * 16]);
      a = wmma_f16(pa, vb, a);
    }
#pragma unroll
    for (int r = 0; r < 8; ++r) {
      const int qi = q0 + r + 8 * mhalf;
      ctx[((size_t)(b * L + qi)) * D + h * DH + 16 * d + nloc] = (h16)a[r];
    }
  }
}

// ---------------------------------------------------------------- GEMM 2: Oproj
// x1pre = ctx @ w_o^T + b_o + tokens(residual), fp32 out; 32M x 64N per wave
__global__ void __launch_bounds__(256)
gemm_oproj_kernel(const h16* __restrict__ A, const h16* __restrict__ W,
                  const float* __restrict__ bias, const float* __restrict__ res,
                  float* __restrict__ out) {
  const int lane = threadIdx.x & 31;
  const int w    = threadIdx.x >> 5;
  const int m0 = blockIdx.x * 128 + (w & 3) * 32;
  const int n0 = blockIdx.y * 128 + (w >> 2) * 64;
  v8f acc[2][4] = {};
  for (int kb = 0; kb < D; kb += 32) {
    v16h a0 = load_a_frag(A, D, m0, kb, lane);
    v16h a1 = load_a_frag(A, D, m0 + 16, kb, lane);
    if (kb + 64 < D) {
      __builtin_prefetch(A + (size_t)(m0 + (lane & 15)) * D + kb + 64, 0, 3);
      __builtin_prefetch(A + (size_t)(m0 + 16 + (lane & 15)) * D + kb + 64, 0, 3);
    }
#pragma unroll
    for (int t = 0; t < 4; ++t) {
      v16h b = load_b_frag(W, D, n0 + 16 * t, kb, lane);
      acc[0][t] = wmma_f16(a0, b, acc[0][t]);
      acc[1][t] = wmma_f16(a1, b, acc[1][t]);
    }
  }
  const int mhalf = lane >> 4, nloc = lane & 15;
#pragma unroll
  for (int t = 0; t < 4; ++t) {
    const int n = n0 + 16 * t + nloc;
    const float bs = bias[n];
#pragma unroll
    for (int mt = 0; mt < 2; ++mt)
#pragma unroll
      for (int r = 0; r < 8; ++r) {
        const int m = m0 + mt * 16 + r + 8 * mhalf;
        const size_t idx = (size_t)m * D + n;
        out[idx] = acc[mt][t][r] + bs + res[idx];
      }
  }
}

// ---------------------------------------------------------------- GEMM 3: SwiGLU
// dual-tile: val & gate accumulated by the same wave; writes f16 val*sigmoid(gate)
// wave tile: 32M x 64N (per side)
__global__ void __launch_bounds__(256)
gemm_ffn1_kernel(const h16* __restrict__ A, const h16* __restrict__ W,
                 const float* __restrict__ bias, h16* __restrict__ g) {
  const int lane = threadIdx.x & 31;
  const int w    = threadIdx.x >> 5;
  const int m0 = blockIdx.x * 128 + (w & 3) * 32;
  const int n0 = blockIdx.y * 128 + (w >> 2) * 64;   // within [0, FF)
  v8f av[2][4] = {}, ag[2][4] = {};
  for (int kb = 0; kb < D; kb += 32) {
    v16h a0 = load_a_frag(A, D, m0, kb, lane);
    v16h a1 = load_a_frag(A, D, m0 + 16, kb, lane);
    if (kb + 64 < D) {
      __builtin_prefetch(A + (size_t)(m0 + (lane & 15)) * D + kb + 64, 0, 3);
      __builtin_prefetch(A + (size_t)(m0 + 16 + (lane & 15)) * D + kb + 64, 0, 3);
    }
#pragma unroll
    for (int t = 0; t < 4; ++t) {
      v16h bv = load_b_frag(W, D, n0 + 16 * t, kb, lane);
      av[0][t] = wmma_f16(a0, bv, av[0][t]);
      av[1][t] = wmma_f16(a1, bv, av[1][t]);
      v16h bg = load_b_frag(W, D, FF + n0 + 16 * t, kb, lane);
      ag[0][t] = wmma_f16(a0, bg, ag[0][t]);
      ag[1][t] = wmma_f16(a1, bg, ag[1][t]);
    }
  }
  const int mhalf = lane >> 4, nloc = lane & 15;
#pragma unroll
  for (int t = 0; t < 4; ++t) {
    const int n = n0 + 16 * t + nloc;
    const float bv = bias[n], bg = bias[FF + n];
#pragma unroll
    for (int mt = 0; mt < 2; ++mt)
#pragma unroll
      for (int r = 0; r < 8; ++r) {
        const int m = m0 + mt * 16 + r + 8 * mhalf;
        const float val  = av[mt][t][r] + bv;
        const float gate = ag[mt][t][r] + bg;
        g[(size_t)m * FF + n] = (h16)(val / (1.0f + __expf(-gate)));
      }
  }
}

// ---------------------------------------------------------------- GEMM 4: FFN out
// y = g @ w_out^T + b_out + x1(residual), fp32; 32M x 64N per wave
__global__ void __launch_bounds__(256)
gemm_ffn2_kernel(const h16* __restrict__ A, const h16* __restrict__ W,
                 const float* __restrict__ bias, const float* __restrict__ res,
                 float* __restrict__ out) {
  const int lane = threadIdx.x & 31;
  const int w    = threadIdx.x >> 5;
  const int m0 = blockIdx.x * 128 + (w & 3) * 32;
  const int n0 = blockIdx.y * 128 + (w >> 2) * 64;
  v8f acc[2][4] = {};
  for (int kb = 0; kb < FF; kb += 32) {
    v16h a0 = load_a_frag(A, FF, m0, kb, lane);
    v16h a1 = load_a_frag(A, FF, m0 + 16, kb, lane);
    if (kb + 64 < FF) {
      __builtin_prefetch(A + (size_t)(m0 + (lane & 15)) * FF + kb + 64, 0, 3);
      __builtin_prefetch(A + (size_t)(m0 + 16 + (lane & 15)) * FF + kb + 64, 0, 3);
    }
#pragma unroll
    for (int t = 0; t < 4; ++t) {
      v16h b = load_b_frag(W, FF, n0 + 16 * t, kb, lane);
      acc[0][t] = wmma_f16(a0, b, acc[0][t]);
      acc[1][t] = wmma_f16(a1, b, acc[1][t]);
    }
  }
  const int mhalf = lane >> 4, nloc = lane & 15;
#pragma unroll
  for (int t = 0; t < 4; ++t) {
    const int n = n0 + 16 * t + nloc;
    const float bs = bias[n];
#pragma unroll
    for (int mt = 0; mt < 2; ++mt)
#pragma unroll
      for (int r = 0; r < 8; ++r) {
        const int m = m0 + mt * 16 + r + 8 * mhalf;
        const size_t idx = (size_t)m * D + n;
        out[idx] = acc[mt][t][r] + bs + res[idx];
      }
  }
}

// ---------------------------------------------------------------- LayerNorms
__device__ __forceinline__ void ln_stats(float v0, float v1, float* mean,
                                         float* rstd, float* red) {
  float s = v0 + v1, ss = v0 * v0 + v1 * v1;
#pragma unroll
  for (int dd = 16; dd >= 1; dd >>= 1) {
    s  += __shfl_xor(s, dd, 32);
    ss += __shfl_xor(ss, dd, 32);
  }
  const int w = threadIdx.x >> 5;
  if ((threadIdx.x & 31) == 0) { red[w] = s; red[w + 8] = ss; }
  __syncthreads();
  s = 0.f; ss = 0.f;
#pragma unroll
  for (int i = 0; i < 8; ++i) { s += red[i]; ss += red[i + 8]; }
  const float m = s * (1.0f / D);
  *mean = m;
  *rstd = rsqrtf(ss * (1.0f / D) - m * m + 1e-5f);
}

__global__ void __launch_bounds__(256)
ln1_kernel(float* __restrict__ x, const float* __restrict__ g,
           const float* __restrict__ b, h16* __restrict__ x16) {
  __shared__ float red[16];
  const size_t base = (size_t)blockIdx.x * D;
  const int i0 = threadIdx.x, i1 = threadIdx.x + 256;
  const float v0 = x[base + i0], v1 = x[base + i1];
  float mean, rstd;
  ln_stats(v0, v1, &mean, &rstd, red);
  const float y0 = (v0 - mean) * rstd * g[i0] + b[i0];
  const float y1 = (v1 - mean) * rstd * g[i1] + b[i1];
  x[base + i0] = y0;  x[base + i1] = y1;
  x16[base + i0] = (h16)y0;  x16[base + i1] = (h16)y1;
}

__global__ void __launch_bounds__(256)
ln2_kernel(const float* __restrict__ x, const float* __restrict__ g,
           const float* __restrict__ b, float* __restrict__ out) {
  __shared__ float red[16];
  const int row = blockIdx.x;
  const size_t base = (size_t)row * D;
  const int i0 = threadIdx.x, i1 = threadIdx.x + 256;
  const float v0 = x[base + i0], v1 = x[base + i1];
  float mean, rstd;
  ln_stats(v0, v1, &mean, &rstd, red);
  const int bb = row >> 7, l = row & (L - 1);
  float* dst = (l == 0) ? out + (size_t)bb * D
                        : out + (size_t)BB * D + ((size_t)bb * S_ + (l - 1)) * D;
  dst[i0] = (v0 - mean) * rstd * g[i0] + b[i0];
  dst[i1] = (v1 - mean) * rstd * g[i1] + b[i1];
}

// ---------------------------------------------------------------- launcher
extern "C" void kernel_launch(void* const* d_in, const int* in_sizes, int n_in,
                              void* d_out, int out_size, void* d_ws, size_t ws_size,
                              hipStream_t stream) {
  const float* node   = (const float*)d_in[0];
  const float* edge   = (const float*)d_in[1];
  const float* cutoff = (const float*)d_in[2];
  const float* w_qkv  = (const float*)d_in[3];
  const float* b_qkv  = (const float*)d_in[4];
  const float* w_o    = (const float*)d_in[5];
  const float* b_o    = (const float*)d_in[6];
  const float* ln1_g  = (const float*)d_in[7];
  const float* ln1_b  = (const float*)d_in[8];
  const float* ln2_g  = (const float*)d_in[9];
  const float* ln2_b  = (const float*)d_in[10];
  const float* w_in   = (const float*)d_in[11];
  const float* b_in   = (const float*)d_in[12];
  const float* w_out  = (const float*)d_in[13];
  const float* b_out  = (const float*)d_in[14];
  float* out = (float*)d_out;

  // workspace carve (aliases noted)
  const size_t SZ32 = (size_t)NT * D * sizeof(float);   // 134 MB
  const size_t SZ16 = (size_t)NT * D * sizeof(h16);     // 67 MB
  char* p = (char*)d_ws;
  float* tok32 = (float*)p;            p += SZ32;   // tokens f32; later y=x1+mlp
  float* x1    = (float*)p;            p += SZ32;   // pre/post LN1 f32
  h16* tok16   = (h16*)p;              p += SZ16;   // tokens f16; later ctx f16
  h16* q16     = (h16*)p;              p += SZ16;   // Q; later gact (low half)
  h16* k16     = (h16*)p;              p += SZ16;   // K; later gact (high half)
  h16* v16     = (h16*)p;              p += SZ16;   // V; later x1 f16
  h16* wq16    = (h16*)p;              p += (size_t)3 * D * D * 2;
  h16* wo16    = (h16*)p;              p += (size_t)D * D * 2;
  h16* wi16    = (h16*)p;              p += (size_t)2 * FF * D * 2;
  h16* wu16    = (h16*)p;              p += (size_t)D * FF * 2;
  h16* ctx16   = tok16;                                // alias
  h16* gact16  = q16;                                  // alias (q||k = NT*FF f16)
  h16* x1f16   = v16;                                  // alias

  dim3 blk(256);
  pack_tokens_kernel<<<4096, blk, 0, stream>>>(node, edge, tok32, tok16);
  cvt_f32_to_f16_kernel<<<512, blk, 0, stream>>>(w_qkv, wq16, 3 * D * D);
  cvt_f32_to_f16_kernel<<<256, blk, 0, stream>>>(w_o, wo16, D * D);
  cvt_f32_to_f16_kernel<<<512, blk, 0, stream>>>(w_in, wi16, 2 * FF * D);
  cvt_f32_to_f16_kernel<<<256, blk, 0, stream>>>(w_out, wu16, D * FF);

  gemm_qkv_kernel<<<dim3(NT / 128, (3 * D) / 128), blk, 0, stream>>>(
      tok16, wq16, b_qkv, q16, k16, v16);
  attn_kernel<<<BB * H_, blk, 0, stream>>>(q16, k16, v16, cutoff, ctx16);
  gemm_oproj_kernel<<<dim3(NT / 128, D / 128), blk, 0, stream>>>(
      ctx16, wo16, b_o, tok32, x1);
  ln1_kernel<<<NT, blk, 0, stream>>>(x1, ln1_g, ln1_b, x1f16);
  gemm_ffn1_kernel<<<dim3(NT / 128, FF / 128), blk, 0, stream>>>(
      x1f16, wi16, b_in, gact16);
  gemm_ffn2_kernel<<<dim3(NT / 128, D / 128), blk, 0, stream>>>(
      gact16, wu16, b_out, x1, tok32);
  ln2_kernel<<<NT, blk, 0, stream>>>(tok32, ln2_g, ln2_b, out);
}
